// Attention_18726057411151
// MI455X (gfx1250) — matrix operand order repaired and checked by reading, not compiled, not run on hardware
//
#include <hip/hip_runtime.h>

// ---------------------------------------------------------------------------
// CDNA5 (gfx1250, wave32) bf16-WMMA attention:
//   x,W -> bf16;  Q=xWq, K=xWk, V=xWv (V stored transposed);
//   flash attention per (b,h) with v_wmma_f32_16x16x32_bf16;
//   out = O @ Wo in fp32.
// Tile staging uses GLOBAL_LOAD_ASYNC_TO_LDS_B128 (ASYNCcnt) with triple-
// buffered LDS: zero staging VGPRs, one barrier per stage.
// ---------------------------------------------------------------------------

typedef __bf16 bf16;
typedef __attribute__((ext_vector_type(16))) __bf16 v16bf;
typedef __attribute__((ext_vector_type(8)))  float  v8f;

#define B_  4
#define S_  2048
#define D_  1024
#define H_  16
#define DK_ 64

union FragU { v16bf v; uint4 q[2]; };

// A-matrix 16x32 bf16 fragment (ISA 7.12.2): lane r=lane&15 holds row M=r.
// half=lane>>4: VGPR0-3 = K[half*8 .. half*8+7], VGPR4-7 = K[16+half*8 ..].
__device__ __forceinline__ v16bf load_frag_a(const bf16* base, int stride, int lane) {
    const int r = lane & 15, half = lane >> 4;
    const bf16* p = base + (size_t)r * stride + half * 8;
    FragU u;
    u.q[0] = *(const uint4*)(p);
    u.q[1] = *(const uint4*)(p + 16);
    return u.v;
}

// B-matrix 32x16 bf16 fragment: lane n=lane&15 holds column N=n.
// half=lane>>4: VGPR0-3 = K[half*8 .. half*8+7], VGPR4-7 = K[16+half*8 ..].
// Load from Bt stored [N][K] row-major (K contiguous).
__device__ __forceinline__ v16bf load_frag_b(const bf16* base, int stride, int lane) {
    const int n = lane & 15, half = lane >> 4;
    const bf16* p = base + (size_t)n * stride + half * 8;
    FragU u;
    u.q[0] = *(const uint4*)(p);
    u.q[1] = *(const uint4*)(p + 16);
    return u.v;
}

__device__ __forceinline__ v8f wmma_bf16(v16bf a, v16bf b, v8f c) {
    return __builtin_amdgcn_wmma_f32_16x16x32_bf16(false, a, false, b, (short)0, c,
                                                   false, false);
}

// ---- CDNA5 async global->LDS copy (no VGPR staging, tracked by ASYNCcnt) ---
// vdst = per-lane LDS byte offset (low 32 bits of generic shared address),
// vaddr = per-lane 64-bit global address. 16 bytes per lane.
__device__ __forceinline__ void async_b128(unsigned int lds_off, const void* gaddr) {
    asm volatile("global_load_async_to_lds_b128 %0, %1, off"
                 :: "v"(lds_off), "v"(gaddr)
                 : "memory");
}
template <int N>
__device__ __forceinline__ void wait_asynccnt() {
    asm volatile("s_wait_asynccnt %0" :: "n"(N) : "memory");
}

// ---------------------------------------------------------------------------
// fp32 -> bf16 conversion, 4 elements/thread
// ---------------------------------------------------------------------------
__global__ void cvt_f32_bf16_x4(const float* __restrict__ in,
                                bf16* __restrict__ out, int n4) {
    int i = blockIdx.x * blockDim.x + threadIdx.x;
    if (i >= n4) return;
    float4 f = reinterpret_cast<const float4*>(in)[i];
    union { bf16 h[4]; uint2 u; } p;
    p.h[0] = (bf16)f.x; p.h[1] = (bf16)f.y; p.h[2] = (bf16)f.z; p.h[3] = (bf16)f.w;
    reinterpret_cast<uint2*>(out)[i] = p.u;
}

// ---------------------------------------------------------------------------
// wt[n][k] = (bf16) w[k][n]  -- LDS tile transpose, block 32x8
// ---------------------------------------------------------------------------
__global__ void transpose_cvt_bf16(const float* __restrict__ w,
                                   bf16* __restrict__ wt, int D) {
    __shared__ float tile[32][33];
    const int n0 = blockIdx.x * 32;
    const int k0 = blockIdx.y * 32;
    const int tx = threadIdx.x, ty = threadIdx.y;
    for (int r = ty; r < 32; r += 8)
        tile[r][tx] = w[(size_t)(k0 + r) * D + n0 + tx];
    __syncthreads();
    for (int r = ty; r < 32; r += 8)
        wt[(size_t)(n0 + r) * D + k0 + tx] = (bf16)tile[tx][r];
}

// ---------------------------------------------------------------------------
// C[8192,1024] = A[8192,1024] * Bt[1024,1024]^T, bf16 in, f32 accum.
// Block: 256 threads = 8 waves (4x2), 128x64 tile; wave tile 32x32; BK=64;
// triple-buffered LDS fed by async copies; 8 WMMAs/wave/stage.
// mode 0: store bf16 into [B,H,S,dk]    (Q, K)
// mode 1: store bf16 into [B,H,dk,S]    (V transposed)
// mode 2: store f32 row-major [M,N]     (final projection)
// ---------------------------------------------------------------------------
#define LDA 72  // padded LDS row stride for BK=64 (multiple of 8 -> b128 aligned)
#define GEMM_K 1024

__global__ void __launch_bounds__(256, 1)
gemm_bf16_wmma(const bf16* __restrict__ A,
               const bf16* __restrict__ Bt,
               void* __restrict__ C, int mode) {
    __shared__ __align__(16) bf16 As[3][128 * LDA];   // 55.3 KB
    __shared__ __align__(16) bf16 Bs[3][64 * LDA];    // 27.6 KB

    const int tid  = threadIdx.x;
    const int wave = tid >> 5;
    const int lane = tid & 31;
    const int bm = blockIdx.x * 128;
    const int bn = blockIdx.y * 64;
    const int mw = (wave >> 1) * 32;   // wave row offset (0..96)
    const int nw = (wave & 1) * 32;    // wave col offset (0,32)

    // staging coords: per stage 4 async b128 of A (rows srow+{0,32,64,96})
    // and 2 of B (rows srow+{0,32}), cols scc..scc+7 of the 64-wide K slab.
    const int srow = tid >> 3;         // 0..31
    const int scc  = (tid & 7) * 8;    // 0..56
    const bf16* aptr = A  + (size_t)(bm + srow) * GEMM_K + scc;
    const bf16* bptr = Bt + (size_t)(bn + srow) * GEMM_K + scc;
    const size_t rstep = (size_t)32 * GEMM_K;

    const unsigned int ABUF = 128 * LDA * sizeof(bf16);   // 18432
    const unsigned int BBUF = 64 * LDA * sizeof(bf16);    //  9216
    const unsigned int RST  = 32 * LDA * sizeof(bf16);    //  4608
    const unsigned int aoff =
        (unsigned int)(uintptr_t)&As[0][0] + (unsigned int)((srow * LDA + scc) * sizeof(bf16));
    const unsigned int boff =
        (unsigned int)(uintptr_t)&Bs[0][0] + (unsigned int)((srow * LDA + scc) * sizeof(bf16));

    // stage 0 -> buffer 0
#pragma unroll
    for (int r = 0; r < 4; ++r) async_b128(aoff + r * RST, aptr + r * rstep);
#pragma unroll
    for (int r = 0; r < 2; ++r) async_b128(boff + r * RST, bptr + r * rstep);
    aptr += 64; bptr += 64;

    v8f acc[2][2] = {};
    const int nsteps = GEMM_K / 64;    // 16
    int buf = 0, nbuf = 1;

    for (int it = 0; it < nsteps; ++it) {
        if (it + 1 < nsteps) {          // issue next stage into buf[nbuf]
#pragma unroll
            for (int r = 0; r < 4; ++r)
                async_b128(aoff + nbuf * ABUF + r * RST, aptr + r * rstep);
#pragma unroll
            for (int r = 0; r < 2; ++r)
                async_b128(boff + nbuf * BBUF + r * RST, bptr + r * rstep);
            aptr += 64; bptr += 64;
            wait_asynccnt<6>();         // current stage's copies have landed
        } else {
            wait_asynccnt<0>();
        }
        __syncthreads();                // all waves' copies of cur are visible

        const bf16* Ac = As[buf];
        const bf16* Bc = Bs[buf];
#pragma unroll
        for (int kk = 0; kk < 64; kk += 32) {
            v16bf af[2], bfg[2];
#pragma unroll
            for (int i = 0; i < 2; ++i)
                af[i] = load_frag_a(&Ac[(mw + i * 16) * LDA + kk], LDA, lane);
#pragma unroll
            for (int j = 0; j < 2; ++j)
                bfg[j] = load_frag_b(&Bc[(nw + j * 16) * LDA + kk], LDA, lane);
#pragma unroll
            for (int i = 0; i < 2; ++i)
#pragma unroll
                for (int j = 0; j < 2; ++j)
                    acc[i][j] = wmma_bf16(af[i], bfg[j], acc[i][j]);
        }
        buf = nbuf;
        nbuf = (nbuf == 2) ? 0 : nbuf + 1;
    }

    // epilogue: C-layout row m = base + g + 8*(lane>>4), col n = base + (lane&15)
    const int half = lane >> 4;
    const int nn   = lane & 15;
#pragma unroll
    for (int i = 0; i < 2; ++i) {
#pragma unroll
        for (int j = 0; j < 2; ++j) {
            const int nloc = bn + nw + j * 16 + nn;
#pragma unroll
            for (int g = 0; g < 8; ++g) {
                const int m = bm + mw + i * 16 + g + 8 * half;
                const float val = acc[i][j][g];
                if (mode == 2) {
                    ((float*)C)[(size_t)m * D_ + nloc] = val;
                } else {
                    const int b = m >> 11, s = m & (S_ - 1);
                    const int h = nloc >> 6, d = nloc & (DK_ - 1);
                    bf16* Cb = (bf16*)C;
                    if (mode == 0)
                        Cb[(((size_t)(b * H_ + h)) * S_ + s) * DK_ + d] = (bf16)val;
                    else
                        Cb[(((size_t)(b * H_ + h)) * DK_ + d) * S_ + s] = (bf16)val;
                }
            }
        }
    }
}

// ---------------------------------------------------------------------------
// Flash attention per (b,h). Block: 256 threads = 8 waves, 128 query rows.
// Each wave owns 16 q-rows; 64x64 K/V tiles are async-copied into triple-
// buffered LDS shared by all 8 waves.
//   Qh [S,64] bf16, Kh [S,64] bf16, Vt [64,S] bf16, O [B,S,D] bf16
// ---------------------------------------------------------------------------
__global__ void __launch_bounds__(256, 1)
flash_attn_wmma(const bf16* __restrict__ Q,
                const bf16* __restrict__ Km,
                const bf16* __restrict__ Vt,
                const int* __restrict__ amask,
                bf16* __restrict__ O) {
    const int b = blockIdx.z, h = blockIdx.y;
    const int tid  = threadIdx.x;
    const int wave = tid >> 5;
    const int lane = tid & 31;
    const int half = lane >> 4;
    const int nn   = lane & 15;
    const int q0   = blockIdx.x * 128 + wave * 16;

    const bf16* Qp = Q  + ((size_t)(b * H_ + h)) * S_ * DK_;
    const bf16* Kp = Km + ((size_t)(b * H_ + h)) * S_ * DK_;
    const bf16* Vp = Vt + ((size_t)(b * H_ + h)) * DK_ * S_;
    const int*  mrow = amask + (size_t)b * S_;

    __shared__ __align__(16) bf16 Ks[3][64 * LDA];    // K tile  [64 k][64 d]
    __shared__ __align__(16) bf16 Vs[3][64 * LDA];    // V tile  [64 d][64 k]
    __shared__ __align__(16) bf16 pbuf[8][16 * LDA];  // per-wave P transpose
    bf16* pb = pbuf[wave];

    const int srow = tid >> 3;         // 0..31
    const int scc  = (tid & 7) * 8;    // 0..56
    const bf16* kptr = Kp + (size_t)srow * DK_ + scc;       // advances by 64*DK_
    const bf16* vptr = Vp + (size_t)srow * S_  + scc;       // advances by 64

    const unsigned int KBUF = 64 * LDA * sizeof(bf16);      // 9216
    const unsigned int RST  = 32 * LDA * sizeof(bf16);      // 4608
    const unsigned int koff =
        (unsigned int)(uintptr_t)&Ks[0][0] + (unsigned int)((srow * LDA + scc) * sizeof(bf16));
    const unsigned int voff =
        (unsigned int)(uintptr_t)&Vs[0][0] + (unsigned int)((srow * LDA + scc) * sizeof(bf16));

    // Q fragments live in registers for the whole pass
    v16bf qf[2];
    qf[0] = load_frag_a(Qp + (size_t)q0 * DK_,      DK_, lane);
    qf[1] = load_frag_a(Qp + (size_t)q0 * DK_ + 32, DK_, lane);

    // stage 0 -> buffer 0
    async_b128(koff,       kptr);
    async_b128(koff + RST, kptr + (size_t)32 * DK_);
    async_b128(voff,       vptr);
    async_b128(voff + RST, vptr + (size_t)32 * S_);
    kptr += (size_t)64 * DK_; vptr += 64;

    v8f oacc[4] = {};
    float mrun[8], lrun[8];
#pragma unroll
    for (int g = 0; g < 8; ++g) { mrun[g] = -__builtin_inff(); lrun[g] = 0.f; }

    const float scale = 0.125f;  // 1/sqrt(64)
    const int ntiles = S_ / 64;  // 32
    int buf = 0, nbuf = 1;

    for (int it = 0; it < ntiles; ++it) {
        const int kt = it * 64;
        if (it + 1 < ntiles) {          // issue next K/V tile into buf[nbuf]
            async_b128(koff + nbuf * KBUF,       kptr);
            async_b128(koff + nbuf * KBUF + RST, kptr + (size_t)32 * DK_);
            async_b128(voff + nbuf * KBUF,       vptr);
            async_b128(voff + nbuf * KBUF + RST, vptr + (size_t)32 * S_);
            kptr += (size_t)64 * DK_; vptr += 64;
            wait_asynccnt<4>();
        } else {
            wait_asynccnt<0>();
        }
        __syncthreads();

        const bf16* Kc = Ks[buf];
        const bf16* Vc = Vs[buf];

        // ---- scores: 16q x 64k via 8 WMMAs -------------------------------
        v8f sc[4] = {};
#pragma unroll
        for (int nj = 0; nj < 4; ++nj) {
            v16bf kb0 = load_frag_b(&Kc[(nj * 16) * LDA],      LDA, lane);
            v16bf kb1 = load_frag_b(&Kc[(nj * 16) * LDA + 32], LDA, lane);
            sc[nj] = wmma_bf16(qf[0], kb0, sc[nj]);
            sc[nj] = wmma_bf16(qf[1], kb1, sc[nj]);
        }

        // ---- scale + key mask -------------------------------------------
#pragma unroll
        for (int nj = 0; nj < 4; ++nj) {
            const float km = (mrow[kt + nj * 16 + nn] != 0) ? 1.f : 0.f;
#pragma unroll
            for (int g = 0; g < 8; ++g)
                sc[nj][g] = km != 0.f ? sc[nj][g] * scale : -__builtin_inff();
        }

        // ---- online softmax per row (row g+8*half lives in 16 lanes) -----
#pragma unroll
        for (int g = 0; g < 8; ++g) {
            float rmax = fmaxf(fmaxf(sc[0][g], sc[1][g]), fmaxf(sc[2][g], sc[3][g]));
#pragma unroll
            for (int d = 8; d >= 1; d >>= 1)
                rmax = fmaxf(rmax, __shfl_xor(rmax, d, 32));
            const float mnew = fmaxf(mrun[g], rmax);
            float corr, rsum = 0.f;
            if (mnew == -__builtin_inff()) {      // nothing unmasked seen yet
                corr = 1.f;
#pragma unroll
                for (int nj = 0; nj < 4; ++nj) sc[nj][g] = 0.f;
            } else {
                corr = __expf(mrun[g] - mnew);
#pragma unroll
                for (int nj = 0; nj < 4; ++nj) {
                    const float e = __expf(sc[nj][g] - mnew);
                    sc[nj][g] = e;
                    rsum += e;
                }
            }
#pragma unroll
            for (int d = 8; d >= 1; d >>= 1)
                rsum += __shfl_xor(rsum, d, 32);
            lrun[g] = lrun[g] * corr + rsum;
            mrun[g] = mnew;
#pragma unroll
            for (int dj = 0; dj < 4; ++dj) oacc[dj][g] *= corr;
        }

        // ---- C-layout -> A-layout transpose of P through per-wave LDS ----
#pragma unroll
        for (int g = 0; g < 8; ++g) {
            const int row = g + 8 * half;
#pragma unroll
            for (int nj = 0; nj < 4; ++nj)
                pb[row * LDA + nj * 16 + nn] = (bf16)sc[nj][g];
        }
        asm volatile("s_wait_dscnt 0" ::: "memory");  // wave-local LDS RAW fence
        v16bf pf0 = load_frag_a(pb,      LDA, lane);
        v16bf pf1 = load_frag_a(pb + 32, LDA, lane);

        // ---- O += P @ V : 8 WMMAs ---------------------------------------
#pragma unroll
        for (int dj = 0; dj < 4; ++dj) {
            v16bf vb0 = load_frag_b(&Vc[(dj * 16) * LDA],      LDA, lane);
            v16bf vb1 = load_frag_b(&Vc[(dj * 16) * LDA + 32], LDA, lane);
            oacc[dj] = wmma_bf16(pf0, vb0, oacc[dj]);
            oacc[dj] = wmma_bf16(pf1, vb1, oacc[dj]);
        }

        buf = nbuf;
        nbuf = (nbuf == 2) ? 0 : nbuf + 1;
    }

    // ---- finalize: /l, row mask (fully-masked rows -> 0), merge heads ----
#pragma unroll
    for (int g = 0; g < 8; ++g) {
        const int s = q0 + g + 8 * half;
        const float inv = (lrun[g] > 0.f && mrow[s] != 0) ? 1.f / lrun[g] : 0.f;
#pragma unroll
        for (int dj = 0; dj < 4; ++dj) {
            const int d = dj * 16 + nn;
            O[((size_t)(b * S_ + s)) * D_ + h * DK_ + d] = (bf16)(oacc[dj][g] * inv);
        }
    }
}

// ---------------------------------------------------------------------------
// host-side orchestration
// ---------------------------------------------------------------------------
extern "C" void kernel_launch(void* const* d_in, const int* in_sizes, int n_in,
                              void* d_out, int out_size, void* d_ws, size_t ws_size,
                              hipStream_t stream) {
    const float* x    = (const float*)d_in[0];
    const int*   am   = (const int*)d_in[1];
    const float* w_q  = (const float*)d_in[2];
    const float* w_k  = (const float*)d_in[3];
    const float* w_v  = (const float*)d_in[4];
    const float* w_o  = (const float*)d_in[5];
    float* out = (float*)d_out;

    const size_t MB = 1u << 20;
    char* ws = (char*)d_ws;
    bf16* xb  = (bf16*)(ws + 0);         // 16 MB  x as bf16 [8192,1024]
    bf16* wqT = (bf16*)(ws + 16 * MB);   //  2 MB  each, [N][K] bf16
    bf16* wkT = (bf16*)(ws + 18 * MB);
    bf16* wvT = (bf16*)(ws + 20 * MB);
    bf16* woT = (bf16*)(ws + 22 * MB);
    bf16* Qb  = (bf16*)(ws + 24 * MB);   // 16 MB  [B,H,S,64]
    bf16* Kb  = (bf16*)(ws + 40 * MB);   // 16 MB  [B,H,S,64]
    bf16* Vtb = (bf16*)(ws + 56 * MB);   // 16 MB  [B,H,64,S]
    bf16* Ob  = (bf16*)(ws + 72 * MB);   // 16 MB  [B,S,D]    (ws >= 88 MB)

    const int M = B_ * S_;               // 8192

    // 1) convert x to bf16
    {
        const int n4 = (M * D_) / 4;     // 2,097,152
        cvt_f32_bf16_x4<<<(n4 + 255) / 256, 256, 0, stream>>>(x, xb, n4);
    }
    // 2) transpose+convert weights
    {
        dim3 g(D_ / 32, D_ / 32), t(32, 8);
        transpose_cvt_bf16<<<g, t, 0, stream>>>(w_q, wqT, D_);
        transpose_cvt_bf16<<<g, t, 0, stream>>>(w_k, wkT, D_);
        transpose_cvt_bf16<<<g, t, 0, stream>>>(w_v, wvT, D_);
        transpose_cvt_bf16<<<g, t, 0, stream>>>(w_o, woT, D_);
    }
    // 3) Q/K/V projections (V stored transposed for the PV WMMA B-operand)
    {
        dim3 g(M / 128, D_ / 64);        // 64 x 16
        gemm_bf16_wmma<<<g, 256, 0, stream>>>(xb, wqT, Qb,  0);
        gemm_bf16_wmma<<<g, 256, 0, stream>>>(xb, wkT, Kb,  0);
        gemm_bf16_wmma<<<g, 256, 0, stream>>>(xb, wvT, Vtb, 1);
    }
    // 4) flash attention
    {
        dim3 g(S_ / 128, H_, B_);        // 16 x 16 x 4
        flash_attn_wmma<<<g, 256, 0, stream>>>(Qb, Kb, Vtb, am, Ob);
    }
    // 5) output projection -> fp32
    {
        dim3 g(M / 128, D_ / 64);
        gemm_bf16_wmma<<<g, 256, 0, stream>>>(Ob, woT, out, 2);
    }
}
